// PTV3_Attention_214748364938
// MI455X (gfx1250) — matrix operand order repaired and checked by reading, not compiled, not run on hardware
//
#include <hip/hip_runtime.h>
#include <hip/hip_bf16.h>

// ---------------------------------------------------------------------------
// PTV3 windowed attention for MI455X (gfx1250, wave32, WMMA).
// [sort] -> [gather + QKV GEMM (bf16 WMMA)] -> [windowed flash attention
// (bf16 WMMA, wave32 shfl softmax, V pre-transposed)] -> [scatter + out-proj].
// Heavy math: v_wmma_f32_16x16x32_bf16. Tile staging: async global->LDS
// (GLOBAL_LOAD_ASYNC_TO_LDS_B128 + s_wait_asynccnt) when available.
// ---------------------------------------------------------------------------

#define G_GRP 4
#define L_GRP 16384
#define N_TOK 65536         // G*L
#define C_DIM 512
#define H_HEAD 8
#define D_HEAD 64
#define P_WIN 512
#define NWIN 128            // N/P

typedef __attribute__((ext_vector_type(16))) __bf16     v16bf;
typedef __attribute__((ext_vector_type(8)))  float      v8f;
typedef __attribute__((ext_vector_type(8)))  unsigned   v8u;
typedef __attribute__((ext_vector_type(4)))  int        v4i_t;

#if defined(__HIP_DEVICE_COMPILE__) && \
    __has_builtin(__builtin_amdgcn_global_load_async_to_lds_b128)
#define USE_ASYNC_LDS 1
#else
#define USE_ASYNC_LDS 0
#endif

#if USE_ASYNC_LDS
// 16B global -> LDS async copy (ASYNCcnt-tracked, no VGPR round trip).
// Builtin signature (per hipcc diagnostic): (int4 AS1*, int4 AS3*, imm, imm).
static __device__ __forceinline__ void async_copy16(const void* g, void* l) {
  __builtin_amdgcn_global_load_async_to_lds_b128(
      (__attribute__((address_space(1))) v4i_t*)(uintptr_t)g,
      (__attribute__((address_space(3))) v4i_t*)(unsigned)(uintptr_t)l, 0, 0);
}
static __device__ __forceinline__ void async_wait0() {
#if __has_builtin(__builtin_amdgcn_s_wait_asynccnt)
  __builtin_amdgcn_s_wait_asynccnt(0);
#else
  asm volatile("s_wait_asynccnt 0x0" ::: "memory");
#endif
}
#endif

static __device__ __forceinline__ unsigned short f2bfu(float f) {
  unsigned u = __builtin_bit_cast(unsigned, f);
  u += 0x7FFFu + ((u >> 16) & 1u);          // round-to-nearest-even
  return (unsigned short)(u >> 16);
}
static __device__ __forceinline__ unsigned pack2bf(float x, float y) {
  return (unsigned)f2bfu(x) | ((unsigned)f2bfu(y) << 16);
}

static __device__ __forceinline__ v8f wmma_bf16(v16bf a, v16bf b, v8f c) {
  return __builtin_amdgcn_wmma_f32_16x16x32_bf16(false, a, false, b,
                                                 (short)0, c, false, false);
}

// A-matrix 16x32 bf16 fragment (row-major tile, lane m=lane%16 is row M,
// half h=lane/16 holds K=16h..16h+15, contiguous, matching B).
static __device__ __forceinline__ v16bf load_a32(const unsigned short* rowptr,
                                                 int h) {
  const unsigned* p = (const unsigned*)rowptr;
  v8u f;
  int hb = h * 8;
#pragma unroll
  for (int j = 0; j < 8; ++j) f[j] = p[hb + j];
  return __builtin_bit_cast(v16bf, f);
}

// B-matrix 32x16 bf16 fragment: lane n=lane%16 is column N; lanes 0-15 hold
// K=0-15, lanes 16-31 hold K=16-31. colptr -> (n, kbase), K contiguous.
static __device__ __forceinline__ v16bf load_b32_contig(
    const unsigned short* colptr, int h) {
  const unsigned* p = (const unsigned*)colptr;
  v8u f;
  int hb = h * 8;
#pragma unroll
  for (int j = 0; j < 8; ++j) f[j] = p[hb + j];
  return __builtin_bit_cast(v16bf, f);
}

// ---------------------------------------------------------------------------
// K1: per-group bitonic argsort of 30-bit codes in 128KB dynamic LDS.
// ---------------------------------------------------------------------------
__global__ __launch_bounds__(1024) void ptv3_sort_kernel(
    const long long* __restrict__ codes, unsigned* __restrict__ perm,
    unsigned* __restrict__ invp) {
  extern __shared__ unsigned smem_sort[];
  unsigned* key = smem_sort;
  unsigned* idx = smem_sort + L_GRP;
  const int g = blockIdx.x, tid = threadIdx.x;

  for (int i = tid; i < L_GRP; i += 1024) {
    key[i] = (unsigned)codes[(size_t)g * L_GRP + i];  // values < 2^30
    idx[i] = (unsigned)i;
  }
  __syncthreads();

  for (int k = 2; k <= L_GRP; k <<= 1) {
    for (int j = k >> 1; j > 0; j >>= 1) {
      for (int i = tid; i < L_GRP; i += 1024) {
        int ixj = i ^ j;
        if (ixj > i) {
          bool up = (i & k) == 0;
          unsigned ka = key[i], kb = key[ixj];
          if ((ka > kb) == up) {
            key[i] = kb; key[ixj] = ka;
            unsigned t = idx[i]; idx[i] = idx[ixj]; idx[ixj] = t;
          }
        }
      }
      __syncthreads();
    }
  }
  for (int i = tid; i < L_GRP; i += 1024) {
    perm[g * L_GRP + i] = (unsigned)(g * L_GRP) + idx[i];
    invp[g * L_GRP + idx[i]] = (unsigned)(g * L_GRP + i);
  }
}

// ---------------------------------------------------------------------------
// K2: gather(perm) + QKV GEMM. 128x128 tiles, K-chunks of 64 staged as bf16.
// q/k: [win][head][p][d] bf16.  v: [win][head][d][p] (pre-transposed for
// contiguous P*V B-fragments in the attention kernel).
// ---------------------------------------------------------------------------
__global__ __launch_bounds__(256) void ptv3_qkv_kernel(
    const float* __restrict__ feats, const float* __restrict__ Wqkv,
    const float* __restrict__ bqkv, const unsigned* __restrict__ perm,
    unsigned short* __restrict__ qws, unsigned short* __restrict__ kws,
    unsigned short* __restrict__ vws) {
  __shared__ unsigned short sX[128 * 64];
  __shared__ unsigned short sW[128 * 64];
  const int bx = blockIdx.x, by = blockIdx.y, tid = threadIdx.x;
  const int w = tid >> 5, lane = tid & 31, h = lane >> 4, ln = lane & 15;

  // Each thread owns half of one tile row for the LDS fills.
  const int frow = tid >> 1, fhalf = tid & 1;
  const unsigned srcrow = perm[bx * 128 + frow];
  const float2* xsrc = (const float2*)(feats + (size_t)srcrow * C_DIM) + fhalf * 16;
  const float2* wsrc =
      (const float2*)(Wqkv + (size_t)(by * 128 + frow) * C_DIM) + fhalf * 16;
  unsigned* dx = (unsigned*)sX + frow * 32 + fhalf * 16;
  unsigned* dw = (unsigned*)sW + frow * 32 + fhalf * 16;

  v8f acc[8] = {};
  for (int kc = 0; kc < 8; ++kc) {
    __syncthreads();
    const float2* xp = xsrc + kc * 32;
    const float2* wp = wsrc + kc * 32;
    __builtin_prefetch(xp + 32, 0, 1);
    __builtin_prefetch(wp + 32, 0, 1);
#pragma unroll
    for (int q = 0; q < 16; ++q) {
      float2 fx = xp[q];
      dx[q] = pack2bf(fx.x, fx.y);
      float2 fw = wp[q];
      dw[q] = pack2bf(fw.x, fw.y);
    }
    __syncthreads();

    v16bf a0 = load_a32(sX + (w * 16 + ln) * 64 + 0, h);
    v16bf a1 = load_a32(sX + (w * 16 + ln) * 64 + 32, h);
    // Software-pipelined N-tile loop: B frags for nt+1 load under WMMA nt.
    v16bf b0 = load_b32_contig(sW + ln * 64 + 0, h);
    v16bf b1 = load_b32_contig(sW + ln * 64 + 32, h);
#pragma unroll
    for (int nt = 0; nt < 8; ++nt) {
      v16bf nb0 = b0, nb1 = b1;
      if (nt < 7) {
        nb0 = load_b32_contig(sW + ((nt + 1) * 16 + ln) * 64 + 0, h);
        nb1 = load_b32_contig(sW + ((nt + 1) * 16 + ln) * 64 + 32, h);
      }
      acc[nt] = wmma_bf16(a0, b0, acc[nt]);
      acc[nt] = wmma_bf16(a1, b1, acc[nt]);
      b0 = nb0; b1 = nb1;
    }
  }

#pragma unroll
  for (int nt = 0; nt < 8; ++nt) {
    int o = by * 128 + nt * 16 + ln;
    float bias = bqkv[o];
    int which = o >> 9, oc = o & 511, hh = oc >> 6, dd = oc & 63;
#pragma unroll
    for (int r = 0; r < 8; ++r) {
      int mg = bx * 128 + w * 16 + r + 8 * h;  // sorted-order row
      int win = mg >> 9, pp = mg & 511;
      unsigned short val = f2bfu(acc[nt][r] + bias);
      if (which == 0) {
        qws[(((size_t)(win * H_HEAD + hh) * P_WIN + pp) * D_HEAD + dd)] = val;
      } else if (which == 1) {
        kws[(((size_t)(win * H_HEAD + hh) * P_WIN + pp) * D_HEAD + dd)] = val;
      } else {  // V stored transposed: [win][head][d][p]
        vws[(((size_t)(win * H_HEAD + hh) * D_HEAD + dd) * P_WIN + pp)] = val;
      }
    }
  }
}

// ---------------------------------------------------------------------------
// K3: windowed flash attention. WG = (window, head, 128-row block); each wave
// owns 16 query rows; streams 16 key-blocks of 32 with online softmax.
// Q/K staged [p][d]; V staged transposed [d][p]; all tiles via async-to-LDS.
// ---------------------------------------------------------------------------
__global__ __launch_bounds__(256) void ptv3_attn_kernel(
    const unsigned short* __restrict__ qws,
    const unsigned short* __restrict__ kws,
    const unsigned short* __restrict__ vws,
    unsigned short* __restrict__ aws) {
  extern __shared__ char smem_attn[];
  unsigned short* sQ = (unsigned short*)smem_attn;  // 128*64  [p][d]
  unsigned short* sK = sQ + 128 * 64;               // 512*64  [p][d]
  unsigned short* sVt = sK + 512 * 64;              // 64*512  [d][p]
  unsigned short* sP = sVt + 64 * 512;              // 8 waves * 16*32

  const int b = blockIdx.x;
  const int win = b >> 5, head = (b >> 2) & 7, rb = b & 3;
  const int tid = threadIdx.x;
  const int w = tid >> 5, lane = tid & 31, h = lane >> 4, ln = lane & 15;

  const size_t kvbase = (size_t)(win * H_HEAD + head) * P_WIN * D_HEAD;
  const size_t qbase = kvbase + (size_t)rb * 128 * D_HEAD;
  const uint4* gq = (const uint4*)(qws + qbase);
  const uint4* gk = (const uint4*)(kws + kvbase);
  const uint4* gv = (const uint4*)(vws + kvbase);  // already [d][p]
  uint4* lq = (uint4*)sQ;
  uint4* lk = (uint4*)sK;
  uint4* lv = (uint4*)sVt;
#if USE_ASYNC_LDS
  for (int e = tid; e < 1024; e += 256) async_copy16(gq + e, lq + e);
  for (int e = tid; e < 4096; e += 256) {
    async_copy16(gk + e, lk + e);
    async_copy16(gv + e, lv + e);
  }
  async_wait0();
#else
  for (int e = tid; e < 1024; e += 256) lq[e] = gq[e];
  for (int e = tid; e < 4096; e += 256) { lk[e] = gk[e]; lv[e] = gv[e]; }
#endif
  __syncthreads();

  const float scale = 0.125f;  // D^-0.5 = 1/8
  v16bf aq0 = load_a32(sQ + (w * 16 + ln) * 64 + 0, h);
  v16bf aq1 = load_a32(sQ + (w * 16 + ln) * 64 + 32, h);

  float m_r[8], l_r[8];
#pragma unroll
  for (int r = 0; r < 8; ++r) { m_r[r] = -__builtin_inff(); l_r[r] = 0.f; }
  v8f oacc[4] = {};
  unsigned short* sPw = sP + w * (16 * 32);  // wave-private P staging

  for (int kb = 0; kb < 16; ++kb) {
    // S = Q * K^T for 32 keys (two 16x16 f32 tiles, K=64 in two WMMAs each)
    v8f c0 = {}, c1 = {};
    {
      v16bf b00 = load_b32_contig(sK + (kb * 32 + ln) * 64 + 0, h);
      v16bf b01 = load_b32_contig(sK + (kb * 32 + ln) * 64 + 32, h);
      v16bf b10 = load_b32_contig(sK + (kb * 32 + 16 + ln) * 64 + 0, h);
      v16bf b11 = load_b32_contig(sK + (kb * 32 + 16 + ln) * 64 + 32, h);
      c0 = wmma_bf16(aq0, b00, c0);
      c0 = wmma_bf16(aq1, b01, c0);
      c1 = wmma_bf16(aq0, b10, c1);
      c1 = wmma_bf16(aq1, b11, c1);
    }
    // online softmax: rows live across 16-lane groups (C layout)
#pragma unroll
    for (int r = 0; r < 8; ++r) {
      float s0 = c0[r] * scale, s1 = c1[r] * scale;
      float mx = fmaxf(s0, s1);
      mx = fmaxf(mx, __shfl_xor(mx, 1));
      mx = fmaxf(mx, __shfl_xor(mx, 2));
      mx = fmaxf(mx, __shfl_xor(mx, 4));
      mx = fmaxf(mx, __shfl_xor(mx, 8));
      float mn = fmaxf(m_r[r], mx);
      float p0 = __expf(s0 - mn), p1 = __expf(s1 - mn);
      float rs = p0 + p1;
      rs += __shfl_xor(rs, 1);
      rs += __shfl_xor(rs, 2);
      rs += __shfl_xor(rs, 4);
      rs += __shfl_xor(rs, 8);
      float al = __expf(m_r[r] - mn);
      l_r[r] = l_r[r] * al + rs;
      m_r[r] = mn;
      oacc[0][r] *= al; oacc[1][r] *= al;
      oacc[2][r] *= al; oacc[3][r] *= al;
      int prow = r + 8 * h;
      sPw[prow * 32 + ln] = f2bfu(p0);
      sPw[prow * 32 + 16 + ln] = f2bfu(p1);
    }
    // O += P * V. V is [d][p] so B-frags are contiguous b128 loads; batch the
    // 4 fragment loads ahead of the 4 WMMAs (DS in-order within a wave).
    v16bf ap = load_a32(sPw + ln * 32, h);
    v16bf bv0 = load_b32_contig(sVt + (0 * 16 + ln) * 512 + kb * 32, h);
    v16bf bv1 = load_b32_contig(sVt + (1 * 16 + ln) * 512 + kb * 32, h);
    v16bf bv2 = load_b32_contig(sVt + (2 * 16 + ln) * 512 + kb * 32, h);
    v16bf bv3 = load_b32_contig(sVt + (3 * 16 + ln) * 512 + kb * 32, h);
    oacc[0] = wmma_bf16(ap, bv0, oacc[0]);
    oacc[1] = wmma_bf16(ap, bv1, oacc[1]);
    oacc[2] = wmma_bf16(ap, bv2, oacc[2]);
    oacc[3] = wmma_bf16(ap, bv3, oacc[3]);
  }

#pragma unroll
  for (int r = 0; r < 8; ++r) {
    float inv_l = 1.0f / l_r[r];
    int row = win * P_WIN + rb * 128 + w * 16 + r + 8 * h;  // sorted order
#pragma unroll
    for (int t = 0; t < 4; ++t) {
      int col = head * D_HEAD + t * 16 + ln;
      aws[(size_t)row * C_DIM + col] = f2bfu(oacc[t][r] * inv_l);
    }
  }
}

// ---------------------------------------------------------------------------
// K4: scatter(inv_perm) + output projection, fp32 out.
// out[n][o] = sum_c attn[inv_perm[n]][c] * Wproj[o][c] + bproj[o]
// attn rows are bf16 already -> async-copy them straight into LDS.
// ---------------------------------------------------------------------------
__global__ __launch_bounds__(256) void ptv3_proj_kernel(
    const unsigned short* __restrict__ aws, const float* __restrict__ Wproj,
    const float* __restrict__ bproj, const unsigned* __restrict__ invp,
    float* __restrict__ out) {
  __shared__ unsigned short sX[128 * 64];
  __shared__ unsigned short sW[128 * 64];
  const int bx = blockIdx.x, by = blockIdx.y, tid = threadIdx.x;
  const int w = tid >> 5, lane = tid & 31, h = lane >> 4, ln = lane & 15;

  const int frow = tid >> 1, fhalf = tid & 1;
  const unsigned srcrow = invp[bx * 128 + frow];
  const unsigned short* xsrc = aws + (size_t)srcrow * C_DIM;
  const float2* wsrc =
      (const float2*)(Wproj + (size_t)(by * 128 + frow) * C_DIM) + fhalf * 16;
  unsigned* dw = (unsigned*)sW + frow * 32 + fhalf * 16;

  v8f acc[8] = {};
  for (int kc = 0; kc < 8; ++kc) {
    __syncthreads();
#if USE_ASYNC_LDS
    {
      const uint4* xg = (const uint4*)(xsrc + kc * 64) + fhalf * 4;
      uint4* xl = (uint4*)sX + frow * 8 + fhalf * 4;
#pragma unroll
      for (int q = 0; q < 4; ++q) async_copy16(xg + q, xl + q);
    }
#else
    {
      const unsigned* xg = (const unsigned*)(xsrc + kc * 64) + fhalf * 16;
      unsigned* xl = (unsigned*)sX + frow * 32 + fhalf * 16;
#pragma unroll
      for (int q = 0; q < 16; ++q) xl[q] = xg[q];
    }
#endif
    const float2* wp = wsrc + kc * 32;
    __builtin_prefetch(wp + 32, 0, 1);
#pragma unroll
    for (int q = 0; q < 16; ++q) {
      float2 fw = wp[q];
      dw[q] = pack2bf(fw.x, fw.y);
    }
#if USE_ASYNC_LDS
    async_wait0();
#endif
    __syncthreads();

    v16bf a0 = load_a32(sX + (w * 16 + ln) * 64 + 0, h);
    v16bf a1 = load_a32(sX + (w * 16 + ln) * 64 + 32, h);
    v16bf b0 = load_b32_contig(sW + ln * 64 + 0, h);
    v16bf b1 = load_b32_contig(sW + ln * 64 + 32, h);
#pragma unroll
    for (int nt = 0; nt < 8; ++nt) {
      v16bf nb0 = b0, nb1 = b1;
      if (nt < 7) {
        nb0 = load_b32_contig(sW + ((nt + 1) * 16 + ln) * 64 + 0, h);
        nb1 = load_b32_contig(sW + ((nt + 1) * 16 + ln) * 64 + 32, h);
      }
      acc[nt] = wmma_bf16(a0, b0, acc[nt]);
      acc[nt] = wmma_bf16(a1, b1, acc[nt]);
      b0 = nb0; b1 = nb1;
    }
  }

#pragma unroll
  for (int nt = 0; nt < 8; ++nt) {
    int o = by * 128 + nt * 16 + ln;
    float bias = bproj[o];
#pragma unroll
    for (int r = 0; r < 8; ++r) {
      int mg = bx * 128 + w * 16 + r + 8 * h;  // original row order
      out[(size_t)mg * C_DIM + o] = acc[nt][r] + bias;
    }
  }
}

// ---------------------------------------------------------------------------
// Host launcher. Workspace layout (~257 MB):
//   [0)       perm      65536 * u32
//   [256KB)   inv_perm  65536 * u32
//   [512KB)   q  bf16 [win][head][p][d]   64 MB
//   [+64MB)   k  bf16 [win][head][p][d]   64 MB
//   [+64MB)   v  bf16 [win][head][d][p]   64 MB   (transposed)
//   [+64MB)   attn bf16 [row][C]          64 MB
// ---------------------------------------------------------------------------
extern "C" void kernel_launch(void* const* d_in, const int* in_sizes, int n_in,
                              void* d_out, int out_size, void* d_ws,
                              size_t ws_size, hipStream_t stream) {
  const float* feats = (const float*)d_in[0];
  const long long* codes = (const long long*)d_in[1];  // int64 per reference
  const float* Wqkv = (const float*)d_in[2];
  const float* bqkv = (const float*)d_in[3];
  const float* Wproj = (const float*)d_in[4];
  const float* bproj = (const float*)d_in[5];
  float* out = (float*)d_out;

  char* ws = (char*)d_ws;
  unsigned* perm = (unsigned*)ws;
  unsigned* invp = perm + N_TOK;
  const size_t half = (size_t)N_TOK * C_DIM;  // elements per bf16 plane
  unsigned short* qws = (unsigned short*)(ws + (1 << 19));
  unsigned short* kws = qws + half;
  unsigned short* vws = kws + half;
  unsigned short* aws = vws + half;

  const int sort_lds = 2 * L_GRP * (int)sizeof(unsigned);           // 128 KB
  const int attn_lds = (128 * 64 + 2 * 512 * 64 + 8 * 16 * 32) * 2; // 152 KB
  (void)hipFuncSetAttribute((const void*)ptv3_sort_kernel,
                            hipFuncAttributeMaxDynamicSharedMemorySize,
                            sort_lds);
  (void)hipFuncSetAttribute((const void*)ptv3_attn_kernel,
                            hipFuncAttributeMaxDynamicSharedMemorySize,
                            attn_lds);

  ptv3_sort_kernel<<<G_GRP, 1024, sort_lds, stream>>>(codes, perm, invp);
  ptv3_qkv_kernel<<<dim3(N_TOK / 128, (3 * C_DIM) / 128), 256, 0, stream>>>(
      feats, Wqkv, bqkv, perm, qws, kws, vws);
  ptv3_attn_kernel<<<dim3(NWIN * H_HEAD * 4), 256, attn_lds, stream>>>(
      qws, kws, vws, aws);
  ptv3_proj_kernel<<<dim3(N_TOK / 128, C_DIM / 128), 256, 0, stream>>>(
      aws, Wproj, bproj, invp, out);
}
